// ASSEMBLY_2370821948029
// MI455X (gfx1250) — matrix operand order repaired and checked by reading, not compiled, not run on hardware
//
#include <hip/hip_runtime.h>

// ---------------------------------------------------------------------------
// DiffPool GNN forward for MI455X (gfx1250, wave32).
// Node-level GCNs: L2-resident gather/scatter with fp32 atomics (bandwidth
// bound, no matrix shape). Pooling contractions (K=1024 batched GEMMs) use
// v_wmma_f32_16x16x32_f16 with LDS tiles stored in *fragment-ready* layout so
// each lane loads its operand fragment with two ds_load_b128 (no u16 gathers).
// ---------------------------------------------------------------------------

#define CDIV(a, b) (((a) + (b) - 1) / (b))

static const int NNODES = 65536;     // B*N
static const int NEDGE  = 2097152;   // B*E
static const int NGRAPH = 64;
static const int NPG    = 1024;      // nodes per graph

typedef __attribute__((ext_vector_type(16))) _Float16 v16h;
typedef __attribute__((ext_vector_type(8)))  float    v8f;

union FAcc { v8f v; float f[8]; };

// ------------------------------- degree ------------------------------------
__global__ void k_fill1(float* p, int n) {
    int i = blockIdx.x * blockDim.x + threadIdx.x;
    if (i < n) p[i] = 1.0f;                        // self-loop weight 1
}
__global__ void k_deg_edge(const int* __restrict__ dst, const float* __restrict__ ew,
                           float* __restrict__ deg, int ne) {
    int i = blockIdx.x * blockDim.x + threadIdx.x;
    if (i < ne) atomicAdd(&deg[dst[i]], ew[i]);
}
__global__ void k_deg_finish(float* p, int n) {
    int i = blockIdx.x * blockDim.x + threadIdx.x;
    if (i < n) {
        float d = p[i];
        p[i] = (d > 0.0f) ? rsqrtf(fmaxf(d, 1e-12f)) : 0.0f;   // dis = deg^-1/2
    }
}

// ------------------------- node-level dense ops ----------------------------
// Y[n,Fout] = X[n,Fin] @ W[Fin,Fout]   (K<=160: VALU, rows broadcast via L0/L2)
__global__ void k_node_gemm(const float* __restrict__ X, const float* __restrict__ W,
                            float* __restrict__ Y, int n, int Fin, int Fout) {
    long long idx = (long long)blockIdx.x * blockDim.x + threadIdx.x;
    long long total = (long long)n * Fout;
    if (idx >= total) return;
    int row = (int)(idx / Fout), col = (int)(idx % Fout);
    const float* xr = X + (long long)row * Fin;
    float a = 0.f;
    for (int k = 0; k < Fin; ++k) a += xr[k] * W[k * Fout + col];
    Y[idx] = a;
}

// out = dis^2 * xw + bias   (self-loop term of GCN)
__global__ void k_gcn_self(const float* __restrict__ XW, const float* __restrict__ dis,
                           const float* __restrict__ bias, float* __restrict__ out,
                           int n, int F) {
    long long idx = (long long)blockIdx.x * blockDim.x + threadIdx.x;
    if (idx >= (long long)n * F) return;
    int row = (int)(idx / F), c = (int)(idx % F);
    float d = dis[row];
    out[idx] = d * d * XW[idx] + bias[c];
}

// one wave (32 lanes) per edge: out[dst,:] += dis[s]*w*dis[d] * xw[src,:]
__global__ void k_gcn_edge(const float* __restrict__ XW, const float* __restrict__ dis,
                           const int* __restrict__ src, const int* __restrict__ dst,
                           const float* __restrict__ ew, float* __restrict__ out,
                           int ne, int F) {
    int t = blockIdx.x * blockDim.x + threadIdx.x;
    int e = t >> 5, lane = t & 31;
    if (e >= ne) return;
    int s = src[e], d = dst[e];
    const float* xs = XW + (long long)s * F;
    __builtin_prefetch(xs, 0, 0);                  // global_prefetch_b8: hide gather
    float nrm = dis[s] * ew[e] * dis[d];
    float* od = out + (long long)d * F;
    for (int c = lane; c < F; c += 32) atomicAdd(&od[c], nrm * xs[c]);
}

// ------------------------------ batchnorm ----------------------------------
// stats: [0,128) = sum, [128,256) = sumsq   (F <= 128)
__global__ void k_bn_stats(const float* __restrict__ X, int rows, int F,
                           float* __restrict__ st) {
    int c = blockIdx.x;
    float s = 0.f, s2 = 0.f;
    for (int r = blockIdx.y * blockDim.x + threadIdx.x; r < rows;
         r += gridDim.y * blockDim.x) {
        float v = X[(long long)r * F + c];
        s += v; s2 += v * v;
    }
    __shared__ float rs[256], rq[256];
    rs[threadIdx.x] = s; rq[threadIdx.x] = s2;
    __syncthreads();
    for (int o = 128; o > 0; o >>= 1) {
        if ((int)threadIdx.x < o) {
            rs[threadIdx.x] += rs[threadIdx.x + o];
            rq[threadIdx.x] += rq[threadIdx.x + o];
        }
        __syncthreads();
    }
    if (threadIdx.x == 0) {
        atomicAdd(&st[c], rs[0]);
        atomicAdd(&st[128 + c], rq[0]);
    }
}
__global__ void k_bn_apply(float* __restrict__ X, const float* __restrict__ st,
                           const float* __restrict__ g, const float* __restrict__ be,
                           int rows, int F) {
    long long idx = (long long)blockIdx.x * blockDim.x + threadIdx.x;
    if (idx >= (long long)rows * F) return;
    int c = (int)(idx % F);
    float inv_n = 1.0f / (float)rows;
    float m = st[c] * inv_n;
    float v = st[128 + c] * inv_n - m * m;
    X[idx] = (X[idx] - m) * rsqrtf(v + 1e-5f) * g[c] + be[c];
}

// ------------------------------ pooling ops --------------------------------
// out[g, f] = max over rows of concat(b0,b1,b2)[g*rows + r, f]
__global__ void k_concat3_maxpool(const float* __restrict__ b0, const float* __restrict__ b1,
                                  const float* __restrict__ b2, int F, int rows,
                                  float* __restrict__ out, int B) {
    int idx = blockIdx.x * blockDim.x + threadIdx.x;
    int total = B * 3 * F;
    if (idx >= total) return;
    int g = idx / (3 * F), f = idx % (3 * F);
    const float* buf = (f < F) ? b0 : ((f < 2 * F) ? b1 : b2);
    int ff = f % F;
    const float* base = buf + ((long long)g * rows) * F + ff;
    float m = base[0];
    for (int r = 1; r < rows; ++r) m = fmaxf(m, base[(long long)r * F]);
    out[idx] = m;
}

// PS[n,100] = [S0|S1|S2] @ Wpfc[160,100] + b
__global__ void k_pfc(const float* __restrict__ S0, const float* __restrict__ S1,
                      const float* __restrict__ S2, const float* __restrict__ W,
                      const float* __restrict__ bias, float* __restrict__ PS, int n) {
    long long idx = (long long)blockIdx.x * blockDim.x + threadIdx.x;
    if (idx >= (long long)n * 100) return;
    int row = (int)(idx / 100), col = (int)(idx % 100);
    float a = bias[col];
    const float* s0 = S0 + (long long)row * 30;
    const float* s1 = S1 + (long long)row * 30;
    const float* s2 = S2 + (long long)row * 100;
    for (int k = 0; k < 30; ++k)  a += s0[k] * W[k * 100 + col];
    for (int k = 0; k < 30; ++k)  a += s1[k] * W[(30 + k) * 100 + col];
    for (int k = 0; k < 100; ++k) a += s2[k] * W[(60 + k) * 100 + col];
    PS[idx] = a;
}

__global__ void k_softmax_rows(float* __restrict__ PS, int n) {
    int r = blockIdx.x * blockDim.x + threadIdx.x;
    if (r >= n) return;
    float* p = PS + (long long)r * 100;
    float mx = p[0];
    for (int i = 1; i < 100; ++i) mx = fmaxf(mx, p[i]);
    float s = 0.f;
    for (int i = 0; i < 100; ++i) { float e = __expf(p[i] - mx); p[i] = e; s += e; }
    float inv = 1.0f / s;
    for (int i = 0; i < 100; ++i) p[i] *= inv;
}

// modularity loss: sum over edges of dot(ps[src], ps[dst]); one wave per edge
__global__ void k_ml_edges(const float* __restrict__ PS, const int* __restrict__ src,
                           const int* __restrict__ dst, float* __restrict__ loss, int ne) {
    int t = blockIdx.x * blockDim.x + threadIdx.x;
    int e = t >> 5, lane = t & 31;
    if (e >= ne) return;
    const float* a = PS + (long long)src[e] * 100;
    const float* b = PS + (long long)dst[e] * 100;
    float p = 0.f;
    for (int c = lane; c < 100; c += 32) p += a[c] * b[c];
    for (int off = 16; off > 0; off >>= 1) p += __shfl_down(p, off, 32);
    if (lane == 0) atomicAdd(loss, p);
}

// entropy loss: sum of -p*log(p+1e-15) over all elements of PS
__global__ void k_entropy(const float* __restrict__ PS, float* __restrict__ loss,
                          long long total) {
    long long i0 = (long long)blockIdx.x * blockDim.x + threadIdx.x;
    long long stride = (long long)gridDim.x * blockDim.x;
    float s = 0.f;
    for (long long i = i0; i < total; i += stride) {
        float p = PS[i];
        s += -p * __logf(p + 1e-15f);
    }
    __shared__ float red[256];
    red[threadIdx.x] = s;
    __syncthreads();
    for (int o = 128; o > 0; o >>= 1) {
        if ((int)threadIdx.x < o) red[threadIdx.x] += red[threadIdx.x + o];
        __syncthreads();
    }
    if (threadIdx.x == 0) atomicAdd(loss, red[0]);
}

// T[b, c, m] += w_e * ps[src_e, c]   for m = dst_e % 1024, b = src_e / 1024.
// (This is t = s^T A without materializing the 268 MB dense adjacency.)
__global__ void k_t_scatter(const float* __restrict__ PS, const int* __restrict__ src,
                            const int* __restrict__ dst, const float* __restrict__ ew,
                            float* __restrict__ T, int ne) {
    int t = blockIdx.x * blockDim.x + threadIdx.x;
    int e = t >> 5, lane = t & 31;
    if (e >= ne) return;
    int s = src[e], d = dst[e];
    int b = s >> 10;          // graph id
    int m = d & 1023;         // dst within graph
    float w = ew[e];
    const float* prow = PS + (long long)s * 100;
    __builtin_prefetch(prow, 0, 0);
    float* tb = T + (long long)b * 102400 + m;
    for (int c = lane; c < 100; c += 32)
        atomicAdd(&tb[(long long)c * 1024], w * prow[c]);
}

// --------------------- WMMA batched GEMM (K = 1024) ------------------------
// D[b] (MxN) = A[b] (MxK) @ B[b] (KxN), fp32 in/out, fp16 WMMA, fp32 accum.
// transA: A element (m,k) read as A[k*lda + m] (used for s^T x).
//
// LDS tiles are stored FRAGMENT-READY: element (m,k) / (k,n) is written at the
// exact (lane, slot) position the WMMA operand layout requires, so each lane
// reads its whole 16-half fragment as one v16h load (2x ds_load_b128).
//   A (16-bit, 16x32): lane = ((k>>3)&1)<<4 | m ; slots follow the 8-block
//     K interleave of the documented A layout.
//   B (16-bit, 32x16): lane = ((k>>3)&1)<<4 | (n&15) ; slot = (k&7) | ((k>>4)<<3).
__launch_bounds__(128)
__global__ void k_wmma_gemm(const float* __restrict__ Ag, const float* __restrict__ Bg,
                            float* __restrict__ Dg, int M, int N, int K,
                            int lda, int ldb, int ldd, int transA,
                            long long strideA, long long strideB, long long strideD) {
    const int b  = blockIdx.y;
    const int mt = blockIdx.x;
    const float* A  = Ag + (long long)b * strideA;
    const float* Bp = Bg + (long long)b * strideB;
    float*       D  = Dg + (long long)b * strideD;

    const int tid  = threadIdx.x;
    const int wave = tid >> 5;
    const int lane = tid & 31;
    const int row  = lane & 15;
    const int hi   = lane >> 4;

    const int ntiles = (N + 15) >> 4;
    const int Npad   = ntiles << 4;     // <= 112

    __shared__ v16h AsF[32];            // A fragments, 1 per lane     (1 KB)
    __shared__ v16h BsF[7 * 32];        // B fragments, per n-tile/lane (7 KB)
    _Float16* Ash = (_Float16*)AsF;
    _Float16* Bsh = (_Float16*)BsF;

    FAcc acc0, acc1;
#pragma unroll
    for (int i = 0; i < 8; ++i) { acc0.f[i] = 0.f; acc1.f[i] = 0.f; }

    for (int k0 = 0; k0 < K; k0 += 32) {
        // stage A tile (16x32) -> fragment layout
        for (int i = tid; i < 16 * 32; i += 128) {
            int m = i >> 5, k = i & 31;
            int gm = (mt << 4) + m;
            float v = 0.f;
            if (gm < M)
                v = transA ? A[(long long)(k0 + k) * lda + gm]
                           : A[(long long)gm * lda + (k0 + k)];
            int hia = (k >> 3) & 1;
            int kb  = k - 8 * hia;                       // {0..7} U {16..23}
            int u   = (kb < 8) ? (kb >> 1) : (4 + ((kb - 16) >> 1));
            int slot = 2 * u + (kb & 1);
            Ash[(((hia << 4) | m) << 4) + slot] = (_Float16)v;
        }
        // stage B tile (32 x Npad) -> fragment layout
        for (int i = tid; i < 32 * Npad; i += 128) {
            int k = i / Npad, nc = i % Npad;
            float v = (nc < N) ? Bp[(long long)(k0 + k) * ldb + nc] : 0.f;
            int lb = (((k >> 3) & 1) << 4) | (nc & 15);
            Bsh[(((nc >> 4) * 32 + lb) << 4) + ((k & 7) | ((k >> 4) << 3))] = (_Float16)v;
        }
        __syncthreads();

        v16h af = AsF[lane];                             // 2x ds_load_b128
        if (wave < ntiles) {                             // wave-uniform (EXEC all-1)
            v16h bf = BsF[wave * 32 + lane];
            acc0.v = __builtin_amdgcn_wmma_f32_16x16x32_f16(
                false, af, false, bf, (short)0, acc0.v, false, false);
        }
        if (wave + 4 < ntiles) {
            v16h bf = BsF[(wave + 4) * 32 + lane];
            acc1.v = __builtin_amdgcn_wmma_f32_16x16x32_f16(
                false, af, false, bf, (short)0, acc1.v, false, false);
        }
        __syncthreads();
    }

    // epilogue: C/D layout — VGPR r holds M = r (lanes 0-15) / r+8 (lanes 16-31)
    if (wave < ntiles) {
        int gn = (wave << 4) + row;
#pragma unroll
        for (int r = 0; r < 8; ++r) {
            int gm = (mt << 4) + r + (hi ? 8 : 0);
            if (gm < M && gn < N) D[(long long)gm * ldd + gn] = acc0.f[r];
        }
    }
    if (wave + 4 < ntiles) {
        int gn = ((wave + 4) << 4) + row;
#pragma unroll
        for (int r = 0; r < 8; ++r) {
            int gm = (mt << 4) + r + (hi ? 8 : 0);
            if (gm < M && gn < N) D[(long long)gm * ldd + gn] = acc1.f[r];
        }
    }
}

// ------------------------- stage-2 dense GCN -------------------------------
// An[b] = dis2 * (p1_adj[b] + I) * dis2   per graph
__global__ void k_stage2_prep(const float* __restrict__ P1A, float* __restrict__ An) {
    int b = blockIdx.x, tid = threadIdx.x;   // 128 threads
    __shared__ float dis2[100];
    const float* Ab = P1A + (long long)b * 10000;
    for (int c = tid; c < 100; c += 128) {
        float s = 1.0f;                        // + I contribution
        for (int d = 0; d < 100; ++d) s += Ab[c * 100 + d];
        dis2[c] = (s > 0.f) ? rsqrtf(fmaxf(s, 1e-12f)) : 0.f;
    }
    __syncthreads();
    for (int i = tid; i < 10000; i += 128) {
        int c = i / 100, d = i % 100;
        float a = Ab[i] + ((c == d) ? 1.0f : 0.0f);
        An[(long long)b * 10000 + i] = dis2[c] * a * dis2[d];
    }
}

// Y[b] = An[b] @ (X[b] @ W) + bias   (X: 100x30, W: 30x30)
__global__ void k_stage2_gcn(const float* __restrict__ X, const float* __restrict__ W,
                             const float* __restrict__ bias, const float* __restrict__ An,
                             float* __restrict__ Y) {
    int b = blockIdx.x, tid = threadIdx.x;   // 256 threads
    __shared__ float XW[100 * 30];
    __shared__ float Ws[30 * 30];
    for (int i = tid; i < 900; i += 256) Ws[i] = W[i];
    __syncthreads();
    const float* Xb = X + (long long)b * 3000;
    for (int i = tid; i < 3000; i += 256) {
        int c = i / 30, f = i % 30;
        float a = 0.f;
        for (int k = 0; k < 30; ++k) a += Xb[c * 30 + k] * Ws[k * 30 + f];
        XW[i] = a;
    }
    __syncthreads();
    const float* Ab = An + (long long)b * 10000;
    for (int i = tid; i < 3000; i += 256) {
        int c = i / 30, f = i % 30;
        float a = bias[f];
        for (int m = 0; m < 100; ++m) a += Ab[c * 100 + m] * XW[m * 30 + f];
        Y[(long long)b * 3000 + i] = a;
    }
}

// ------------------------------- MLP head ----------------------------------
__global__ void k_head(const float* __restrict__ conv, const float* __restrict__ x2o,
                       const float* __restrict__ w1, const float* __restrict__ b1,
                       const float* __restrict__ w2, const float* __restrict__ b2,
                       float* __restrict__ out) {
    int b = blockIdx.x, tid = threadIdx.x;   // 64 threads
    __shared__ float fe[180];
    __shared__ float hh[50];
    for (int i = tid; i < 90; i += 64) {
        fe[i]      = conv[b * 90 + i];
        fe[90 + i] = x2o[b * 90 + i];
    }
    __syncthreads();
    for (int j = tid; j < 50; j += 64) {
        float a = b1[j];
        for (int k = 0; k < 180; ++k) a += fe[k] * w1[k * 50 + j];
        hh[j] = fmaxf(a, 0.f);
    }
    __syncthreads();
    for (int k = tid; k < 6; k += 64) {
        float a = b2[k];
        for (int q = 0; q < 50; ++q) a += hh[q] * w2[q * 6 + k];
        out[b * 6 + k] = a;
    }
}

__global__ void k_loss_finish(const float* __restrict__ L, float* __restrict__ out) {
    if (blockIdx.x == 0 && threadIdx.x == 0)
        out[384] = -L[0] * (1.0f / 2097152.0f) + L[1] * (1.0f / 65536.0f);
}

// ===========================================================================
extern "C" void kernel_launch(void* const* d_in, const int* in_sizes, int n_in,
                              void* d_out, int out_size, void* d_ws, size_t ws_size,
                              hipStream_t stream) {
    (void)in_sizes; (void)n_in; (void)out_size; (void)ws_size;
    const int n = NNODES, ne = NEDGE;

    const float* x   = (const float*)d_in[0];
    const float* pos = (const float*)d_in[1];
    const float* ew  = (const float*)d_in[2];
    const int*  eidx = (const int*)d_in[3];
    const int*  src  = eidx;
    const int*  dst  = eidx + ne;
#define PRM(i) ((const float*)d_in[(i)])

    // ---- workspace carve-out (~126 MB of fp32) ----
    float* ws   = (float*)d_ws;
    float* dis  = ws;                                  // n
    float* A0   = dis + n;                             // n*30 (x11)
    float* A1   = A0 + (size_t)n * 30;                 // n*30 (x12)
    float* A2   = A1 + (size_t)n * 30;                 // n*30 (x13)
    float* S0   = A2 + (size_t)n * 30;                 // n*30 (s11)
    float* S1   = S0 + (size_t)n * 30;                 // n*30 (s12)
    float* S2   = S1 + (size_t)n * 30;                 // n*100 (s13; reused as T)
    float* PS   = S2 + (size_t)n * 100;                // n*100 (assignments)
    float* SCR  = PS + (size_t)n * 100;                // n*100 (per-layer xw)
    float* P1X  = SCR + (size_t)n * 100;               // 64*100*30
    float* P1A  = P1X + 192000;                        // 64*100*100
    float* AN   = P1A + 640000;                        // 64*100*100
    float* Y1   = AN + 640000;                         // 64*100*30
    float* Y2   = Y1 + 192000;
    float* Y3   = Y2 + 192000;
    float* CONV = Y3 + 192000;                         // 64*90
    float* X2O  = CONV + 5760;                         // 64*90
    float* STATS = X2O + 5760;                         // 256 (BN sum/sumsq)
    float* LOSS  = STATS + 256;                        // 2 (ml-sum, el-sum)

    const int ewave = (ne * 32) / 256;                 // wave-per-edge grid

    // ---- symmetric-norm degrees ----
    k_fill1<<<CDIV(n, 256), 256, 0, stream>>>(dis, n);
    k_deg_edge<<<CDIV(ne, 256), 256, 0, stream>>>(dst, ew, dis, ne);
    k_deg_finish<<<CDIV(n, 256), 256, 0, stream>>>(dis, n);

    // ---- one GCN + BN layer on 65536 nodes ----
    auto gcn_layer = [&](const float* in, int Fin, int wi, int bi,
                         float* out, int F, int gi, int bei) {
        int tot = n * F;
        k_node_gemm<<<CDIV(tot, 256), 256, 0, stream>>>(in, PRM(wi), SCR, n, Fin, F);
        k_gcn_self<<<CDIV(tot, 256), 256, 0, stream>>>(SCR, dis, PRM(bi), out, n, F);
        k_gcn_edge<<<ewave, 256, 0, stream>>>(SCR, dis, src, dst, ew, out, ne, F);
        hipMemsetAsync(STATS, 0, 256 * sizeof(float), stream);
        k_bn_stats<<<dim3(F, 32), 256, 0, stream>>>(out, n, F, STATS);
        k_bn_apply<<<CDIV(tot, 256), 256, 0, stream>>>(out, STATS, PRM(gi), PRM(bei), n, F);
    };

    // x tower: c11/c12/c13 + n11/n12/n13
    gcn_layer(x,  3, 5, 6,  A0, 30, 29, 30);
    gcn_layer(A0, 30, 7, 8,  A1, 30, 31, 32);
    gcn_layer(A1, 30, 9, 10, A2, 30, 33, 34);
    // pos tower: p11/p12/p13 + np11/np12/np13
    gcn_layer(pos, 44, 11, 12, S0, 30, 35, 36);
    gcn_layer(S0,  30, 13, 14, S1, 30, 37, 38);
    gcn_layer(S1,  30, 15, 16, S2, 100, 39, 40);

    // per-graph max pool of [x11|x12|x13]
    k_concat3_maxpool<<<CDIV(NGRAPH * 90, 256), 256, 0, stream>>>(
        A0, A1, A2, 30, NPG, CONV, NGRAPH);

    // assignments: ps = softmax([s11|s12|s13] @ pfc)
    k_pfc<<<CDIV(n * 100, 256), 256, 0, stream>>>(S0, S1, S2, PRM(23), PRM(24), PS, n);
    k_softmax_rows<<<CDIV(n, 256), 256, 0, stream>>>(PS, n);

    // losses
    hipMemsetAsync(LOSS, 0, 2 * sizeof(float), stream);
    k_ml_edges<<<ewave, 256, 0, stream>>>(PS, src, dst, LOSS, ne);
    k_entropy<<<4096, 256, 0, stream>>>(PS, LOSS + 1, (long long)n * 100);

    // t = s^T A via edge scatter (reuses S2: s13 no longer needed)
    float* T = S2;
    hipMemsetAsync(T, 0, (size_t)n * 100 * sizeof(float), stream);
    k_t_scatter<<<ewave, 256, 0, stream>>>(PS, src, dst, ew, T, ne);

    // p1_x[b] = ps[b]^T (100x1024) @ x13[b] (1024x30)   — WMMA, transA
    k_wmma_gemm<<<dim3(7, NGRAPH), 128, 0, stream>>>(
        PS, A2, P1X, 100, 30, 1024,
        /*lda=*/100, /*ldb=*/30, /*ldd=*/30, /*transA=*/1,
        /*strideA=*/(long long)NPG * 100, /*strideB=*/(long long)NPG * 30,
        /*strideD=*/3000);
    // p1_adj[b] = t[b] (100x1024) @ ps[b] (1024x100)    — WMMA
    k_wmma_gemm<<<dim3(7, NGRAPH), 128, 0, stream>>>(
        T, PS, P1A, 100, 100, 1024,
        /*lda=*/1024, /*ldb=*/100, /*ldd=*/100, /*transA=*/0,
        /*strideA=*/102400LL, /*strideB=*/(long long)NPG * 100,
        /*strideD=*/10000LL);

    // stage-2 dense GCN tower on pooled graphs
    k_stage2_prep<<<NGRAPH, 128, 0, stream>>>(P1A, AN);

    auto gcn2_layer = [&](const float* in, int wi, int bi, float* out, int gi, int bei) {
        k_stage2_gcn<<<NGRAPH, 256, 0, stream>>>(in, PRM(wi), PRM(bi), AN, out);
        hipMemsetAsync(STATS, 0, 256 * sizeof(float), stream);
        k_bn_stats<<<dim3(30, 4), 256, 0, stream>>>(out, NGRAPH * 100, 30, STATS);
        k_bn_apply<<<CDIV(NGRAPH * 100 * 30, 256), 256, 0, stream>>>(
            out, STATS, PRM(gi), PRM(bei), NGRAPH * 100, 30);
    };
    gcn2_layer(P1X, 17, 18, Y1, 41, 42);
    gcn2_layer(Y1,  19, 20, Y2, 43, 44);
    gcn2_layer(Y2,  21, 22, Y3, 45, 46);

    k_concat3_maxpool<<<CDIV(NGRAPH * 90, 256), 256, 0, stream>>>(
        Y1, Y2, Y3, 30, 100, X2O, NGRAPH);

    // classifier head + loss
    k_head<<<NGRAPH, 64, 0, stream>>>(CONV, X2O, PRM(25), PRM(26), PRM(27), PRM(28),
                                      (float*)d_out);
    k_loss_finish<<<1, 1, 0, stream>>>(LOSS, (float*)d_out);
#undef PRM
}
